// GenResNet_84782654423196
// MI455X (gfx1250) — matrix operand order repaired and checked by reading, not compiled, not run on hardware
//
#include <hip/hip_runtime.h>
#include <cstdint>

// ---------- types ----------
typedef __attribute__((ext_vector_type(16))) __bf16 v16bf;
typedef __attribute__((ext_vector_type(8)))  float  v8f;
typedef __attribute__((ext_vector_type(4)))  unsigned int u32x4;
typedef __attribute__((ext_vector_type(4)))  float  f32x4;

union Frag16 {
  u32x4 u[2];
  v16bf v;
};

static __device__ __forceinline__ void atomic_add_f32(float* p, float v) {
  __hip_atomic_fetch_add(p, v, __ATOMIC_RELAXED, __HIP_MEMORY_SCOPE_AGENT);
}

// ---------- GEMM: C[M,N] = A[M,K](bf16,row-major) x Bt[N,K](bf16, B transposed) ----------
// One wave computes a 16x64 tile via 4x v_wmma_f32_16x16x32_bf16 per 32-wide k-step.
__global__ __launch_bounds__(256) void gemm_bf16_wmma(
    const __bf16* __restrict__ A, const __bf16* __restrict__ Bt,
    float* __restrict__ C, int M, int N, int K) {
  const int lane = threadIdx.x & 31;
  const int wave = threadIdx.x >> 5;
  const int ngroups = N >> 6;                       // 64-column groups
  const int gw = blockIdx.x * (blockDim.x >> 5) + wave;
  const int mt = gw / ngroups;
  const int ng = gw - mt * ngroups;
  if (mt * 16 >= M) return;                         // wave-uniform: EXEC stays all-1 for live waves
  const int m0 = mt << 4;
  const int n0 = ng << 6;
  const int l15 = lane & 15;
  const int hi  = lane >> 4;

  // A fragment (16x32): lanes 0-15 row m0+l, K {k0..k0+15};
  // lanes 16-31 row m0+l, K {k0+16..k0+31} (contiguous, matches B).
  const __bf16* Ap = A + (size_t)(m0 + l15) * K + (hi << 4);
  // B fragment (32x16): lane holds column n0+l15 (+16j), K {k0 + hi*16 .. +15} contiguous in Bt.
  const __bf16* Bp = Bt + (size_t)(n0 + l15) * K + (hi << 4);

  v8f acc0 = {}, acc1 = {}, acc2 = {}, acc3 = {};
  for (int k0 = 0; k0 < K; k0 += 32) {
    Frag16 a, b0, b1, b2, b3;
    a.u[0] = *(const u32x4*)(Ap + k0);
    a.u[1] = *(const u32x4*)(Ap + k0 + 8);
    const __bf16* bp = Bp + k0;
    b0.u[0] = *(const u32x4*)(bp);
    b0.u[1] = *(const u32x4*)(bp + 8);
    b1.u[0] = *(const u32x4*)(bp + 16 * (size_t)K);
    b1.u[1] = *(const u32x4*)(bp + 16 * (size_t)K + 8);
    b2.u[0] = *(const u32x4*)(bp + 32 * (size_t)K);
    b2.u[1] = *(const u32x4*)(bp + 32 * (size_t)K + 8);
    b3.u[0] = *(const u32x4*)(bp + 48 * (size_t)K);
    b3.u[1] = *(const u32x4*)(bp + 48 * (size_t)K + 8);
    acc0 = __builtin_amdgcn_wmma_f32_16x16x32_bf16(false, a.v, false, b0.v, (short)0, acc0, false, false);
    acc1 = __builtin_amdgcn_wmma_f32_16x16x32_bf16(false, a.v, false, b1.v, (short)0, acc1, false, false);
    acc2 = __builtin_amdgcn_wmma_f32_16x16x32_bf16(false, a.v, false, b2.v, (short)0, acc2, false, false);
    acc3 = __builtin_amdgcn_wmma_f32_16x16x32_bf16(false, a.v, false, b3.v, (short)0, acc3, false, false);
  }

  // C/D layout: VGPR i -> row m0 + hi*8 + i, col n0 + l15 (+16 per tile).
  const int rb = m0 + (hi << 3);
  float* Cp = C + (size_t)rb * N + n0 + l15;
  #pragma unroll
  for (int i = 0; i < 8; ++i) {
    Cp[(size_t)i * N]      = acc0[i];
    Cp[(size_t)i * N + 16] = acc1[i];
    Cp[(size_t)i * N + 32] = acc2[i];
    Cp[(size_t)i * N + 48] = acc3[i];
  }
}

// ---------- helpers ----------
__global__ void zero_f32_kernel(float* __restrict__ p, int n) {
  int t = blockIdx.x * blockDim.x + threadIdx.x;
  if (t < n) p[t] = 0.0f;
}

__global__ void f32_to_bf16_kernel(const float* __restrict__ in, __bf16* __restrict__ out, size_t n) {
  size_t t = (size_t)blockIdx.x * blockDim.x + threadIdx.x;
  if (t < n) out[t] = (__bf16)in[t];
}

// W: [K, Ncols] f32  ->  Wt: [Ncols, K] bf16
__global__ void transpose_f32_to_bf16_kernel(const float* __restrict__ W, __bf16* __restrict__ Wt,
                                             int K, int Ncols) {
  int t = blockIdx.x * blockDim.x + threadIdx.x;
  if (t >= K * Ncols) return;
  int n = t / K;
  int k = t - n * K;
  Wt[t] = (__bf16)W[(size_t)k * Ncols + n];
}

// Column sums / sums-of-squares. blockDim.x == C. Y row stride ld, column offset coff.
__global__ void colstats_kernel(const float* __restrict__ Y, int ld, int coff, int C,
                                int rows, int rowsPerBlock, float* __restrict__ stats) {
  int c = threadIdx.x;
  int r0 = blockIdx.x * rowsPerBlock;
  int r1 = r0 + rowsPerBlock; if (r1 > rows) r1 = rows;
  float s = 0.f, s2 = 0.f;
  for (int r = r0; r < r1; ++r) {
    float v = Y[(size_t)r * ld + coff + c];
    s += v; s2 += v * v;
  }
  atomic_add_f32(&stats[c], s);
  atomic_add_f32(&stats[C + c], s2);
}

// ab[c] = gamma*rsqrt(var+eps); ab[C+c] = beta - mean*scale
__global__ void bn_finalize_kernel(const float* __restrict__ stats, const float* __restrict__ gamma,
                                   const float* __restrict__ beta, int C, float invN,
                                   float* __restrict__ ab) {
  int c = blockIdx.x * blockDim.x + threadIdx.x;
  if (c >= C) return;
  float mean = stats[c] * invN;
  float var  = stats[C + c] * invN - mean * mean;
  float sc   = gamma[c] * rsqrtf(var + 1e-5f);
  ab[c] = sc;
  ab[C + c] = beta[c] - mean * sc;
}

// out_bf16[t] = bf16(relu(Y[r*ld+coff+c]*a[c]+b[c])), c = t & cmask, r = t >> cshift
__global__ void bn_relu_bf16_kernel(const float* __restrict__ Y, int ld, int coff,
                                    const float* __restrict__ ab, int C, int cmask, int cshift,
                                    size_t total, __bf16* __restrict__ out) {
  size_t t = (size_t)blockIdx.x * blockDim.x + threadIdx.x;
  if (t >= total) return;
  int c = (int)(t & (size_t)cmask);
  size_t r = t >> cshift;
  float v = fmaf(Y[r * ld + coff + c], ab[c], ab[C + c]);
  out[t] = (__bf16)fmaxf(v, 0.0f);
}

// Per (edge, 4-channel) scatter: y2[dst, 128+c] += y2[src, c]  (n -> agg into r region)
__global__ void scatter_add_kernel(const long long* __restrict__ ei, int E, float* __restrict__ y2) {
  int t = blockIdx.x * blockDim.x + threadIdx.x;
  int e  = t >> 5;
  int c4 = t & 31;
  if (e >= E) return;
  long long s = ei[e];
  long long d = ei[(size_t)E + e];
  f32x4 m = *(const f32x4*)(y2 + (size_t)s * 256 + (size_t)c4 * 4);
  float* dst = y2 + (size_t)d * 256 + 128 + (size_t)c4 * 4;
  atomic_add_f32(dst + 0, m.x);
  atomic_add_f32(dst + 1, m.y);
  atomic_add_f32(dst + 2, m.z);
  atomic_add_f32(dst + 3, m.w);
}

// out = relu(Y3*a + b + x), C = 512
__global__ void bn_residual_relu_kernel(const float* __restrict__ Y3, const float* __restrict__ x,
                                        const float* __restrict__ ab, float* __restrict__ out,
                                        size_t total) {
  size_t t = (size_t)blockIdx.x * blockDim.x + threadIdx.x;
  if (t >= total) return;
  int c = (int)(t & 511);
  float v = fmaf(Y3[t], ab[c], ab[512 + c]) + x[t];
  out[t] = fmaxf(v, 0.0f);
}

// ---------- launcher ----------
extern "C" void kernel_launch(void* const* d_in, const int* in_sizes, int n_in,
                              void* d_out, int out_size, void* d_ws, size_t ws_size,
                              hipStream_t stream) {
  const int CIN = 512, CH = 128, CC = 256, COUT = 512;
  const int Nn = in_sizes[0] / CIN;           // 50000 nodes
  const int E  = in_sizes[1] / 2;             // 800000 edges

  const float*     x  = (const float*)d_in[0];
  const long long* ei = (const long long*)d_in[1];
  const float* W1 = (const float*)d_in[2];
  const float* g1 = (const float*)d_in[3];
  const float* b1 = (const float*)d_in[4];
  const float* Ws = (const float*)d_in[5];
  const float* Wn = (const float*)d_in[6];
  const float* g2 = (const float*)d_in[7];
  const float* b2 = (const float*)d_in[8];
  const float* W3 = (const float*)d_in[9];
  const float* g3 = (const float*)d_in[10];
  const float* b3 = (const float*)d_in[11];
  float* out = (float*)d_out;

  // ---- workspace layout (lifetime-aliased) ----
  char* w = (char*)d_ws;
  const size_t sz_y3 = (size_t)Nn * COUT * sizeof(float);     // 102.4 MB
  const size_t sz_y2 = (size_t)Nn * CC * sizeof(float);       // 51.2 MB
  const size_t sz_h  = (size_t)Nn * CH * sizeof(__bf16);      // 12.8 MB
  float*  y3  = (float*)w;                                    // [0, sz_y3)
  __bf16* xb  = (__bf16*)w;                                   // aliases y3 (dead before gemm3)
  float*  y1  = (float*)(w + sz_y3);                          // aliases y2 region
  float*  y2  = (float*)(w + sz_y3);
  __bf16* hb  = (__bf16*)(w + sz_y3 + sz_y2);                 // h, later aliased by h2
  __bf16* h2b = hb;
  char* wts = w + sz_y3 + sz_y2 + sz_h;
  __bf16* W1t = (__bf16*)wts;                                 // [128][512]
  __bf16* Wct = (__bf16*)(wts + (size_t)CH * CIN * 2);        // [256][128]  (Wn cols | Ws cols)
  __bf16* W3t = (__bf16*)(wts + (size_t)CH * CIN * 2 + (size_t)CC * CH * 2); // [512][128]
  float* stats1 = (float*)(wts + (size_t)CH * CIN * 2 + (size_t)CC * CH * 2 + (size_t)COUT * CH * 2);
  float* ab1    = stats1 + 2 * CH;
  float* stats2 = ab1 + 2 * CH;
  float* ab2    = stats2 + 2 * CH;
  float* stats3 = ab2 + 2 * CH;
  float* ab3    = stats3 + 2 * COUT;

  const float invN = 1.0f / (float)Nn;

  // 0) zero BN stat accumulators (ab* fully overwritten, stats need zeroing)
  zero_f32_kernel<<<(2 * (CH + CH + COUT) + 255) / 256, 256, 0, stream>>>(stats1, 2 * (CH + CH + COUT) + 4 * CH);
  // (stats1..ab3 are contiguous; count is 2048 but the 6-block grid zeroes only the first 1536 floats)

  // 1) convert x -> bf16; weights -> transposed bf16
  {
    size_t n = (size_t)Nn * CIN;
    f32_to_bf16_kernel<<<(unsigned)((n + 255) / 256), 256, 0, stream>>>(x, xb, n);
  }
  transpose_f32_to_bf16_kernel<<<(CIN * CH + 255) / 256, 256, 0, stream>>>(W1, W1t, CIN, CH);
  transpose_f32_to_bf16_kernel<<<(CH * CH + 255) / 256, 256, 0, stream>>>(Wn, Wct, CH, CH);
  transpose_f32_to_bf16_kernel<<<(CH * CH + 255) / 256, 256, 0, stream>>>(Ws, Wct + (size_t)CH * CH, CH, CH);
  transpose_f32_to_bf16_kernel<<<(CH * COUT + 255) / 256, 256, 0, stream>>>(W3, W3t, CH, COUT);

  // 2) GEMM1: y1 = x @ W1   [Nn,512]x[512,128]
  {
    int waves = (Nn / 16) * (CH / 64);
    gemm_bf16_wmma<<<(waves + 7) / 8, 256, 0, stream>>>(xb, W1t, y1, Nn, CH, CIN);
  }
  // 3) BN1 stats + finalize + apply -> h (bf16)
  colstats_kernel<<<(Nn + 127) / 128, CH, 0, stream>>>(y1, CH, 0, CH, Nn, 128, stats1);
  bn_finalize_kernel<<<1, CH, 0, stream>>>(stats1, g1, b1, CH, invN, ab1);
  {
    size_t n = (size_t)Nn * CH;
    bn_relu_bf16_kernel<<<(unsigned)((n + 255) / 256), 256, 0, stream>>>(y1, CH, 0, ab1, CH, CH - 1, 7, n, hb);
  }
  // 4) GEMM2 (fused): y2 = h @ [Wn | Ws]  -> cols 0..127 = n, 128..255 = r
  {
    int waves = (Nn / 16) * (CC / 64);
    gemm_bf16_wmma<<<(waves + 7) / 8, 256, 0, stream>>>(hb, Wct, y2, Nn, CC, CH);
  }
  // 5) scatter-add: y2[dst, 128+c] += y2[src, c]  (L2-resident atomics)
  {
    size_t n = (size_t)E * 32;
    scatter_add_kernel<<<(unsigned)((n + 255) / 256), 256, 0, stream>>>(ei, E, y2);
  }
  // 6) BN2 on r region + apply -> h2 (bf16)
  colstats_kernel<<<(Nn + 127) / 128, CH, 0, stream>>>(y2, CC, CH, CH, Nn, 128, stats2);
  bn_finalize_kernel<<<1, CH, 0, stream>>>(stats2, g2, b2, CH, invN, ab2);
  {
    size_t n = (size_t)Nn * CH;
    bn_relu_bf16_kernel<<<(unsigned)((n + 255) / 256), 256, 0, stream>>>(y2, CC, CH, ab2, CH, CH - 1, 7, n, h2b);
  }
  // 7) GEMM3: y3 = h2 @ W3   [Nn,128]x[128,512]
  {
    int waves = (Nn / 16) * (COUT / 64);
    gemm_bf16_wmma<<<(waves + 7) / 8, 256, 0, stream>>>(h2b, W3t, y3, Nn, COUT, CH);
  }
  // 8) BN3 stats + finalize + residual + relu -> out
  colstats_kernel<<<(Nn + 127) / 128, COUT, 0, stream>>>(y3, COUT, 0, COUT, Nn, 128, stats3);
  bn_finalize_kernel<<<(COUT + 255) / 256, 256, 0, stream>>>(stats3, g3, b3, COUT, invN, ab3);
  {
    size_t n = (size_t)Nn * COUT;
    bn_residual_relu_kernel<<<(unsigned)((n + 255) / 256), 256, 0, stream>>>(y3, x, ab3, out, n);
  }
  (void)n_in; (void)out_size; (void)ws_size;
}
